// ChebyKANFeedForward_72610717106404
// MI455X (gfx1250) — matrix operand order repaired and checked by reading, not compiled, not run on hardware
//
#include <hip/hip_runtime.h>
#include <hip/hip_bf16.h>

#define DIM_IN   1024
#define HIDDEN   4096
#define NDEG     5        // degree + 1
#define TOKENS   8192
#define LN_EPS   1e-5f

typedef __attribute__((ext_vector_type(16))) __bf16 bfrag16;
typedef __attribute__((ext_vector_type(8)))  __bf16 bfrag8;
typedef __attribute__((ext_vector_type(8)))  float  ffrag8;

// ---------------------------------------------------------------------------
// Prep: elementwise tanh
// ---------------------------------------------------------------------------
__global__ __launch_bounds__(256) void k_tanh(const float* __restrict__ x,
                                              float* __restrict__ xt, int n) {
  int i = blockIdx.x * 256 + threadIdx.x;
  if (i < n) xt[i] = tanhf(x[i]);
}

// ---------------------------------------------------------------------------
// Prep: repack coeffs [in, out, NDEG] (f32) -> Bt [out, K] (bf16), K = d*in + i
// ---------------------------------------------------------------------------
__global__ __launch_bounds__(256) void k_repack_coeffs(const float* __restrict__ c,
                                                       __bf16* __restrict__ bt,
                                                       int in_dim, int out_dim) {
  size_t idx = (size_t)blockIdx.x * 256 + threadIdx.x;
  size_t total = (size_t)in_dim * out_dim * NDEG;
  if (idx >= total) return;
  int K = in_dim * NDEG;
  int o = (int)(idx / K);
  int k = (int)(idx % K);
  int d = k / in_dim;
  int i = k % in_dim;
  bt[(size_t)o * K + k] = (__bf16)c[((size_t)i * out_dim + o) * NDEG + d];
}

// ---------------------------------------------------------------------------
// Fused Chebyshev-basis GEMM, double-buffered, async B staging.
//   out[M=8192, N] (f32) = T_d(xt)[M, K] x Bt[N, K]^T,  K = NDEG*IN
// Block: 256 threads = 8 wave32s; tile 128(M) x 128(N); K-stage 64.
// Wave grid 2(M) x 4(N); each wave owns 64x32 C -> acc[4][2] v8f.
// Per stage: 16 v_wmma_f32_16x16x32_bf16.
// B tile staged with GLOBAL_LOAD_ASYNC_TO_LDS_B128 (ASYNCcnt), A tile computed
// (Chebyshev on tanh-values) and ds_store'd, both into the "next" buffer while
// WMMA consumes "cur".
// ---------------------------------------------------------------------------
template <int IN>
__global__ __launch_bounds__(256)
void k_cheby_gemm(const float* __restrict__ xt, const __bf16* __restrict__ bt,
                  float* __restrict__ out, int N) {
  constexpr int K      = NDEG * IN;
  constexpr int SPD    = IN / 64;        // stages per degree (16 or 64; pow2)
  constexpr int NSTAGE = NDEG * SPD;

  __shared__ __align__(16) __bf16 As[2][128][72];   // 128x64 + pad, x2 buffers
  __shared__ __align__(16) __bf16 Bs[2][128][72];

  const int tid  = threadIdx.x;
  const int lane = tid & 31;
  const int wv   = tid >> 5;
  const int wm   = wv >> 2;        // 0..1 -> M offset wm*64
  const int wn   = wv & 3;         // 0..3 -> N offset wn*32
  const int hsel = lane >> 4;      // lane half: 0/1
  const int l16  = lane & 15;

  const int n0 = blockIdx.x * 128;
  const int m0 = blockIdx.y * 128;

  ffrag8 acc[4][2];
#pragma unroll
  for (int mi = 0; mi < 4; ++mi)
#pragma unroll
    for (int ni = 0; ni < 2; ++ni)
#pragma unroll
      for (int e = 0; e < 8; ++e) acc[mi][ni][e] = 0.0f;

  // staging map: 128 rows x 64 cols, 32 elems (64B) per thread
  const int srow = tid >> 1;            // 0..127
  const int scol = (tid & 1) * 32;      // 0 or 32

  const float*  aBase = &xt[(size_t)(m0 + srow) * IN + scol];
  const __bf16* bBase = &bt[(size_t)(n0 + srow) * K + scol];

  // ---- B stage: 4x async 16B global->LDS copies (ASYNCcnt tracked) ----
  auto stageB = [&](int s, int buf) {
    const __bf16* bp = bBase + ((size_t)s << 6);
    unsigned lds = (unsigned)(uintptr_t)&Bs[buf][srow][scol];
    asm volatile("global_load_async_to_lds_b128 %0, %1, off"
                 :: "v"(lds), "v"(bp) : "memory");
    asm volatile("global_load_async_to_lds_b128 %0, %1, off offset:16"
                 :: "v"(lds), "v"(bp) : "memory");
    asm volatile("global_load_async_to_lds_b128 %0, %1, off offset:32"
                 :: "v"(lds), "v"(bp) : "memory");
    asm volatile("global_load_async_to_lds_b128 %0, %1, off offset:48"
                 :: "v"(lds), "v"(bp) : "memory");
  };

  // ---- A stage: T_d(tanh-values) f32 -> bf16 -> ds_store ----
  auto stageA = [&](int s, int buf) {
    const int d = s / SPD;                               // shift (SPD pow2)
    const float* ap = aBase + ((size_t)(s % SPD) << 6);  // mask
    float4 t[8];
#pragma unroll
    for (int q = 0; q < 8; ++q) t[q] = ((const float4*)ap)[q];
    const float* tv = (const float*)t;
    __align__(16) __bf16 a[32];
#pragma unroll
    for (int e = 0; e < 32; ++e) {
      float xv = tv[e];
      float r;
      if (d == 0)      r = 1.0f;
      else if (d == 1) r = xv;
      else if (d == 2) r = 2.0f * xv * xv - 1.0f;
      else if (d == 3) r = xv * (4.0f * xv * xv - 3.0f);
      else { float x2 = xv * xv; r = 8.0f * x2 * (x2 - 1.0f) + 1.0f; }
      a[e] = (__bf16)r;
    }
#pragma unroll
    for (int q = 0; q < 4; ++q)
      *(bfrag8*)&As[buf][srow][scol + q * 8] = *(const bfrag8*)&a[q * 8];
  };

  // wait for async B copies + LDS stores, then workgroup barrier
#define ASYNC_BARRIER()                                   \
  do {                                                    \
    asm volatile("s_wait_asynccnt 0x0" ::: "memory");     \
    __syncthreads();                                      \
  } while (0)

  // ---- prologue: fill buffer 0 ----
  stageB(0, 0);
  stageA(0, 0);
  ASYNC_BARRIER();

  // ---- pipelined main loop ----
  for (int s = 0; s < NSTAGE; ++s) {
    const int cur = s & 1;
    if (s + 1 < NSTAGE) {               // stage next buffer while computing
      stageB(s + 1, cur ^ 1);
      stageA(s + 1, cur ^ 1);
    }

#pragma unroll
    for (int ks = 0; ks < 64; ks += 32) {
      bfrag16 afr[4], bfr[2];
#pragma unroll
      for (int mi = 0; mi < 4; ++mi) {
        int row = wm * 64 + mi * 16 + l16;                          // M = l16
        bfrag8 lo = *(const bfrag8*)&As[cur][row][ks + hsel * 8];   // K=8h+0..7
        bfrag8 hi = *(const bfrag8*)&As[cur][row][ks + 16 + hsel * 8];
#pragma unroll
        for (int e = 0; e < 8; ++e) { afr[mi][e] = lo[e]; afr[mi][8 + e] = hi[e]; }
      }
#pragma unroll
      for (int ni = 0; ni < 2; ++ni) {
        int row = wn * 32 + ni * 16 + l16;                          // N = l16
        bfrag8 lo = *(const bfrag8*)&Bs[cur][row][ks + hsel * 8];
        bfrag8 hi = *(const bfrag8*)&Bs[cur][row][ks + 16 + hsel * 8];
#pragma unroll
        for (int e = 0; e < 8; ++e) { bfr[ni][e] = lo[e]; bfr[ni][8 + e] = hi[e]; }
      }
#pragma unroll
      for (int mi = 0; mi < 4; ++mi)
#pragma unroll
        for (int ni = 0; ni < 2; ++ni)
          acc[mi][ni] = __builtin_amdgcn_wmma_f32_16x16x32_bf16(
              false, afr[mi], false, bfr[ni], (short)0, acc[mi][ni], false, false);
    }

    ASYNC_BARRIER();                    // next buffer complete + visible
  }
#undef ASYNC_BARRIER

  // ---- epilogue: C layout (VGPR j -> M = 8*hsel + j, N = l16) ----
#pragma unroll
  for (int mi = 0; mi < 4; ++mi) {
#pragma unroll
    for (int ni = 0; ni < 2; ++ni) {
      int col = n0 + wn * 32 + ni * 16 + l16;
#pragma unroll
      for (int j = 0; j < 8; ++j) {
        int row = m0 + wm * 64 + mi * 16 + hsel * 8 + j;
        out[(size_t)row * N + col] = acc[mi][ni][j];
      }
    }
  }
}

// ---------------------------------------------------------------------------
// Row LayerNorm, in place; optionally fuses tanh (prep for next cheby layer)
// ---------------------------------------------------------------------------
__global__ __launch_bounds__(256)
void k_layernorm(float* __restrict__ h, const float* __restrict__ gamma,
                 const float* __restrict__ beta, int N, int apply_tanh) {
  __shared__ float red[256];
  const int tid = threadIdx.x;
  float* p = h + (size_t)blockIdx.x * N;
  const int npt = N / 256;               // <= 16

  float v[16];
  float s = 0.0f;
  for (int j = 0; j < npt; ++j) { v[j] = p[tid + j * 256]; s += v[j]; }
  red[tid] = s;
  __syncthreads();
  for (int off = 128; off > 0; off >>= 1) {
    if (tid < off) red[tid] += red[tid + off];
    __syncthreads();
  }
  float mu = red[0] / (float)N;
  __syncthreads();

  float sq = 0.0f;
  for (int j = 0; j < npt; ++j) { float dd = v[j] - mu; sq += dd * dd; }
  red[tid] = sq;
  __syncthreads();
  for (int off = 128; off > 0; off >>= 1) {
    if (tid < off) red[tid] += red[tid + off];
    __syncthreads();
  }
  float rstd = rsqrtf(red[0] / (float)N + LN_EPS);
  __syncthreads();

  for (int j = 0; j < npt; ++j) {
    int c = tid + j * 256;
    float o = (v[j] - mu) * rstd * gamma[c] + beta[c];
    p[c] = apply_tanh ? tanhf(o) : o;
  }
}

// ---------------------------------------------------------------------------
// Launch
// ---------------------------------------------------------------------------
extern "C" void kernel_launch(void* const* d_in, const int* in_sizes, int n_in,
                              void* d_out, int out_size, void* d_ws, size_t ws_size,
                              hipStream_t stream) {
  (void)in_sizes; (void)n_in; (void)out_size; (void)ws_size;

  const float* x  = (const float*)d_in[0];   // [8192, 1024]
  const float* c1 = (const float*)d_in[1];   // [1024, 4096, 5]
  const float* g1 = (const float*)d_in[2];
  const float* b1 = (const float*)d_in[3];
  const float* c2 = (const float*)d_in[4];   // [4096, 1024, 5]
  const float* g2 = (const float*)d_in[5];
  const float* b2 = (const float*)d_in[6];
  float* out = (float*)d_out;                // [8192, 1024]

  // workspace layout (all 16B-aligned offsets)
  char* ws = (char*)d_ws;
  __bf16* Bt1 = (__bf16*)(ws);                               //  4096 x  5120 bf16 (40 MiB)
  __bf16* Bt2 = (__bf16*)(ws + 41943040ull);                 //  1024 x 20480 bf16 (40 MiB)
  float*  XT1 = (float*)(ws + 83886080ull);                  //  8192 x  1024 f32  (32 MiB)
  float*  H1  = (float*)(ws + 117440512ull);                 //  8192 x  4096 f32 (128 MiB)

  // 1) tanh of input
  k_tanh<<<(TOKENS * DIM_IN) / 256, 256, 0, stream>>>(x, XT1, TOKENS * DIM_IN);

  // 2) repack coeffs to bf16 [N, K]
  k_repack_coeffs<<<((size_t)DIM_IN * HIDDEN * NDEG + 255) / 256, 256, 0, stream>>>(
      c1, Bt1, DIM_IN, HIDDEN);
  k_repack_coeffs<<<((size_t)HIDDEN * DIM_IN * NDEG + 255) / 256, 256, 0, stream>>>(
      c2, Bt2, HIDDEN, DIM_IN);

  // 3) layer 1: basis GEMM  [8192,5120] x [5120,4096] -> H1
  {
    dim3 grid(HIDDEN / 128, TOKENS / 128);
    k_cheby_gemm<DIM_IN><<<grid, 256, 0, stream>>>(XT1, Bt1, H1, HIDDEN);
  }

  // 4) LN1 fused with tanh -> H1 now holds tanh(LN(h)) for layer 2
  k_layernorm<<<TOKENS, 256, 0, stream>>>(H1, g1, b1, HIDDEN, 1);

  // 5) layer 2: basis GEMM  [8192,20480] x [20480,1024] -> d_out
  {
    dim3 grid(DIM_IN / 128, TOKENS / 128);
    k_cheby_gemm<HIDDEN><<<grid, 256, 0, stream>>>(H1, Bt2, out, DIM_IN);
  }

  // 6) final LN in place on d_out
  k_layernorm<<<TOKENS, 256, 0, stream>>>(out, g2, b2, DIM_IN, 0);
}
